// BankedDenoiser_70600672412083
// MI455X (gfx1250) — compile-verified, matrix operand order checked by reading, not run on hardware
//
#include <hip/hip_runtime.h>
#include <hip/hip_bf16.h>
#include <math.h>

// ---------------------------------------------------------------------------
// BankedDenoiser forward for gfx1250 (MI455X).
// B=16 S=512 IN=64 D=512 H=8 L=4 DFF=2048 M=1024 K=4 DH=64
// All matmuls: v_wmma_f32_16x16x32_bf16. All WMMA operands are pre-converted
// bf16, [rows, K] K-contiguous, so fragment loads are global_load_b128.
// ---------------------------------------------------------------------------

typedef __attribute__((ext_vector_type(16))) __bf16 v16bf;
typedef __attribute__((ext_vector_type(8)))  __bf16 bf16x8;
typedef __attribute__((ext_vector_type(8)))  float  v8f;

union V16 { v16bf v; bf16x8 h[2]; };

static __device__ __forceinline__ v8f vzero8() {
  v8f z;
#pragma unroll
  for (int i = 0; i < 8; ++i) z[i] = 0.0f;
  return z;
}

// CDNA5 16-bit fragment K map: VGPR r<4 -> K = 2r + 8*half (+k0),
// r>=4 -> K = 16 + 2(r-4) + 8*half. So h[0] = 8 bf16 @ k0+8*half,
// h[1] = 8 bf16 @ k0+16+8*half: two 16B loads per fragment.
static __device__ __forceinline__ int kbase(int r, int half) {
  return ((r < 4) ? (2 * r) : (16 + 2 * (r - 4))) + 8 * half;
}

static __device__ __forceinline__ v16bf load_frag(const __bf16* row, int k0, int half) {
  V16 u;
  u.h[0] = *(const bf16x8*)(row + k0 + 8 * half);
  u.h[1] = *(const bf16x8*)(row + k0 + 16 + 8 * half);
  return u.v;
}

static __device__ __forceinline__ v8f wmma_bf16(const v16bf& a, const v16bf& b, const v8f& c) {
  return __builtin_amdgcn_wmma_f32_16x16x32_bf16(false, a, false, b, (short)0, c, false, false);
}

// ---------------------------------------------------------------------------
// WMMA GEMM, bf16 x bf16 -> f32 acc. A:[M,K], B:[N,K] (both K-contiguous).
// 4 waves / block, each wave one 16x16 tile (4 adjacent N tiles).
// EPI: 0 none | 1 +bias[n] | 2 +bias,relu | 3 embed(bias+PE+t_embed)
//      4 *scale | 5 bank prior: -0.3*(sq[m]+sq[n]-2*acc)+log(Size[n])
// WF: write f32 C ; WB: write bf16 Cb
// ---------------------------------------------------------------------------
template <int EPI, int WF, int WB>
__global__ __launch_bounds__(128) void gemm_bf16_wmma(
    const __bf16* __restrict__ A, const __bf16* __restrict__ B,
    const float* __restrict__ bias, float* __restrict__ C,
    __bf16* __restrict__ Cb, int ntn, int Kd, int lda, int ldb, int ldc,
    float scale, const float* __restrict__ e0, const float* __restrict__ e1) {
  const int lane = threadIdx.x & 31;
  const int tile = (int)blockIdx.x * 4 + (threadIdx.x >> 5);
  const int lo = lane & 15, half = lane >> 4;
  const int tm = tile / ntn, tn = tile % ntn;
  const int m0 = tm << 4, n0 = tn << 4;
  v8f acc = vzero8();
  const __bf16* arow = A + (size_t)(m0 + lo) * lda;
  const __bf16* brow = B + (size_t)(n0 + lo) * ldb;
  for (int k0 = 0; k0 < Kd; k0 += 32) {
    if (k0 + 32 < Kd) {  // next-chunk prefetch -> global_prefetch_b8
      __builtin_prefetch(arow + k0 + 32 + 8 * half, 0, 1);
      __builtin_prefetch(brow + k0 + 32 + 8 * half, 0, 1);
    }
    const v16bf a = load_frag(arow, k0, half);
    const v16bf b = load_frag(brow, k0, half);
    acc = wmma_bf16(a, b, acc);
  }
#pragma unroll
  for (int r = 0; r < 8; ++r) {
    const int m = m0 + r + 8 * half;
    const int n = n0 + lo;
    float v = acc[r];
    if (EPI == 1 || EPI == 2) v += bias[n];
    if (EPI == 2) v = fmaxf(v, 0.0f);
    if (EPI == 3) {  // bias + sinusoidal PE + t_embed (S=512, D=512)
      v += bias[n];
      const int s = m & 511, bb = m >> 9;
      const float ang = (float)s * __expf(-(float)(n & ~1) * 0.017988946f);  // ln(1e4)/512
      v += ((n & 1) ? __cosf(ang) : __sinf(ang)) + e0[(bb << 9) + n];
    }
    if (EPI == 4) v *= scale;
    if (EPI == 5) v = -0.3f * (e0[m] + e0[n] - 2.0f * v) + __logf(e1[n]);
    if (WF) C[(size_t)m * ldc + n] = v;
    if (WB) Cb[(size_t)m * ldc + n] = (__bf16)v;
  }
}

// ---------------------------------------------------------------------------
// Token attention (flash-style): one wave per (b, head, 16-query tile).
// qkv packed bf16 [B*S, 3D]; q@0, k@D, v@2D; head h at h*64. Output bf16.
// ---------------------------------------------------------------------------
__global__ __launch_bounds__(32) void token_attn(const __bf16* __restrict__ qkv,
                                                 __bf16* __restrict__ o) {
  __shared__ __bf16 pls[16 * 32];
  const int lane = threadIdx.x, lo = lane & 15, half = lane >> 4;
  const int qt = blockIdx.x & 31;          // S/16
  const int hh = (blockIdx.x >> 5) & 7;    // H
  const int b  = blockIdx.x >> 8;          // B
  const int LDQ = 1536;
  const __bf16* base = qkv + (size_t)b * 512 * LDQ;
  const int q0 = qt << 4;

  v16bf aQ[2];
  {
    const __bf16* qrow = base + (size_t)(q0 + lo) * LDQ + hh * 64;
    aQ[0] = load_frag(qrow, 0, half);
    aQ[1] = load_frag(qrow, 32, half);
  }
  float mrow[8], lrow[8];
  v8f Oacc[4];
#pragma unroll
  for (int r = 0; r < 8; ++r) { mrow[r] = -3.0e38f; lrow[r] = 0.0f; }
#pragma unroll
  for (int d = 0; d < 4; ++d) Oacc[d] = vzero8();

  for (int j0 = 0; j0 < 512; j0 += 32) {
    v8f s0 = vzero8(), s1 = vzero8();
    const __bf16* krow0 = base + (size_t)(j0 + lo) * LDQ + 512 + hh * 64;
    const __bf16* krow1 = krow0 + (size_t)16 * LDQ;
#pragma unroll
    for (int kc = 0; kc < 2; ++kc) {
      s0 = wmma_bf16(aQ[kc], load_frag(krow0, kc * 32, half), s0);
      s1 = wmma_bf16(aQ[kc], load_frag(krow1, kc * 32, half), s1);
    }
    // scale + online softmax (row stats replicated across each 16-lane half)
#pragma unroll
    for (int r = 0; r < 8; ++r) {
      s0[r] *= 0.125f;  // 1/sqrt(64)
      s1[r] *= 0.125f;
      float t = fmaxf(s0[r], s1[r]);
      t = fmaxf(t, __shfl_xor(t, 1, 32));
      t = fmaxf(t, __shfl_xor(t, 2, 32));
      t = fmaxf(t, __shfl_xor(t, 4, 32));
      t = fmaxf(t, __shfl_xor(t, 8, 32));
      const float nm = fmaxf(mrow[r], t);
      const float corr = __expf(mrow[r] - nm);
      mrow[r] = nm;
      const float p0 = __expf(s0[r] - nm);
      const float p1 = __expf(s1[r] - nm);
      s0[r] = p0; s1[r] = p1;
      float bs = p0 + p1;
      bs += __shfl_xor(bs, 1, 32);
      bs += __shfl_xor(bs, 2, 32);
      bs += __shfl_xor(bs, 4, 32);
      bs += __shfl_xor(bs, 8, 32);
      lrow[r] = lrow[r] * corr + bs;
#pragma unroll
      for (int d = 0; d < 4; ++d) Oacc[d][r] *= corr;
    }
    // P: C-layout -> A-fragment via LDS
#pragma unroll
    for (int r = 0; r < 8; ++r) {
      const int row = r + 8 * half;
      pls[row * 32 + lo]      = (__bf16)s0[r];
      pls[row * 32 + 16 + lo] = (__bf16)s1[r];
    }
    __syncthreads();
    v16bf aP;
#pragma unroll
    for (int r = 0; r < 8; ++r) {
      const int kb = kbase(r, half);
      aP[2 * r]     = pls[lo * 32 + kb];
      aP[2 * r + 1] = pls[lo * 32 + kb + 1];
    }
    __syncthreads();
#pragma unroll
    for (int d = 0; d < 4; ++d) {
      v16bf bV;
      const __bf16* vcol = base + 1024 + hh * 64 + d * 16 + lo;
#pragma unroll
      for (int r = 0; r < 8; ++r) {
        const int kb = kbase(r, half);
        bV[2 * r]     = vcol[(size_t)(j0 + kb) * LDQ];
        bV[2 * r + 1] = vcol[(size_t)(j0 + kb + 1) * LDQ];
      }
      Oacc[d] = wmma_bf16(aP, bV, Oacc[d]);
    }
  }
#pragma unroll
  for (int d = 0; d < 4; ++d)
#pragma unroll
    for (int r = 0; r < 8; ++r) {
      const int m = q0 + r + 8 * half;
      o[((size_t)b * 512 + m) * 512 + hh * 64 + d * 16 + lo] =
          (__bf16)(Oacc[d][r] / lrow[r]);
    }
}

// ---------------------------------------------------------------------------
// Bank attention: score = dot/8 + prior[i,j]; softmax over j; out = P*bv.
// One wave per (head, 16-bank tile). bq/bk/bv bf16 [1024,512]; out bf16.
// ---------------------------------------------------------------------------
__global__ __launch_bounds__(32) void bank_attn(
    const __bf16* __restrict__ bq, const __bf16* __restrict__ bk,
    const __bf16* __restrict__ bv, const float* __restrict__ prior,
    __bf16* __restrict__ z) {
  __shared__ __bf16 pls[16 * 32];
  const int lane = threadIdx.x, lo = lane & 15, half = lane >> 4;
  const int it = blockIdx.x & 63;   // M/16
  const int hh = blockIdx.x >> 6;   // H
  const int i0 = it << 4;

  v16bf aQ[2];
  {
    const __bf16* qrow = bq + (size_t)(i0 + lo) * 512 + hh * 64;
    aQ[0] = load_frag(qrow, 0, half);
    aQ[1] = load_frag(qrow, 32, half);
  }
  float mrow[8], lrow[8];
  v8f Oacc[4];
#pragma unroll
  for (int r = 0; r < 8; ++r) { mrow[r] = -3.0e38f; lrow[r] = 0.0f; }
#pragma unroll
  for (int d = 0; d < 4; ++d) Oacc[d] = vzero8();

  for (int j0 = 0; j0 < 1024; j0 += 32) {
    v8f s0 = vzero8(), s1 = vzero8();
    const __bf16* krow0 = bk + (size_t)(j0 + lo) * 512 + hh * 64;
    const __bf16* krow1 = krow0 + (size_t)16 * 512;
#pragma unroll
    for (int kc = 0; kc < 2; ++kc) {
      s0 = wmma_bf16(aQ[kc], load_frag(krow0, kc * 32, half), s0);
      s1 = wmma_bf16(aQ[kc], load_frag(krow1, kc * 32, half), s1);
    }
#pragma unroll
    for (int r = 0; r < 8; ++r) {
      const int i = i0 + r + 8 * half;
      s0[r] = s0[r] * 0.125f + prior[(size_t)i * 1024 + j0 + lo];
      s1[r] = s1[r] * 0.125f + prior[(size_t)i * 1024 + j0 + 16 + lo];
    }
#pragma unroll
    for (int r = 0; r < 8; ++r) {
      float t = fmaxf(s0[r], s1[r]);
      t = fmaxf(t, __shfl_xor(t, 1, 32));
      t = fmaxf(t, __shfl_xor(t, 2, 32));
      t = fmaxf(t, __shfl_xor(t, 4, 32));
      t = fmaxf(t, __shfl_xor(t, 8, 32));
      const float nm = fmaxf(mrow[r], t);
      const float corr = __expf(mrow[r] - nm);
      mrow[r] = nm;
      const float p0 = __expf(s0[r] - nm);
      const float p1 = __expf(s1[r] - nm);
      s0[r] = p0; s1[r] = p1;
      float bs = p0 + p1;
      bs += __shfl_xor(bs, 1, 32);
      bs += __shfl_xor(bs, 2, 32);
      bs += __shfl_xor(bs, 4, 32);
      bs += __shfl_xor(bs, 8, 32);
      lrow[r] = lrow[r] * corr + bs;
#pragma unroll
      for (int d = 0; d < 4; ++d) Oacc[d][r] *= corr;
    }
#pragma unroll
    for (int r = 0; r < 8; ++r) {
      const int row = r + 8 * half;
      pls[row * 32 + lo]      = (__bf16)s0[r];
      pls[row * 32 + 16 + lo] = (__bf16)s1[r];
    }
    __syncthreads();
    v16bf aP;
#pragma unroll
    for (int r = 0; r < 8; ++r) {
      const int kb = kbase(r, half);
      aP[2 * r]     = pls[lo * 32 + kb];
      aP[2 * r + 1] = pls[lo * 32 + kb + 1];
    }
    __syncthreads();
#pragma unroll
    for (int d = 0; d < 4; ++d) {
      v16bf bV;
      const __bf16* vcol = bv + hh * 64 + d * 16 + lo;
#pragma unroll
      for (int r = 0; r < 8; ++r) {
        const int kb = kbase(r, half);
        bV[2 * r]     = vcol[(size_t)(j0 + kb) * 512];
        bV[2 * r + 1] = vcol[(size_t)(j0 + kb + 1) * 512];
      }
      Oacc[d] = wmma_bf16(aP, bV, Oacc[d]);
    }
  }
#pragma unroll
  for (int d = 0; d < 4; ++d)
#pragma unroll
    for (int r = 0; r < 8; ++r) {
      const int m = i0 + r + 8 * half;
      z[(size_t)m * 512 + hh * 64 + d * 16 + lo] = (__bf16)(Oacc[d][r] / lrow[r]);
    }
}

// ---------------------------------------------------------------------------
// out = LayerNorm(base + delta) * g + b ; wave per 512-row; dual f32+bf16 out.
// ---------------------------------------------------------------------------
__global__ __launch_bounds__(256) void residual_ln(
    const float* __restrict__ base, const float* __restrict__ delta,
    const float* __restrict__ g, const float* __restrict__ bta,
    float* __restrict__ out, __bf16* __restrict__ outb) {
  const int row = blockIdx.x * 8 + (threadIdx.x >> 5);
  const int lane = threadIdx.x & 31;
  const float* pb = base + (size_t)row * 512;
  const float* pd = delta + (size_t)row * 512;
  float vals[16];
  float s = 0.0f;
#pragma unroll
  for (int i = 0; i < 16; ++i) {
    const int c = lane + 32 * i;
    const float x = pb[c] + pd[c];
    vals[i] = x; s += x;
  }
#pragma unroll
  for (int off = 16; off > 0; off >>= 1) s += __shfl_xor(s, off, 32);
  const float mu = s * (1.0f / 512.0f);
  float vs = 0.0f;
#pragma unroll
  for (int i = 0; i < 16; ++i) { const float d = vals[i] - mu; vs += d * d; }
#pragma unroll
  for (int off = 16; off > 0; off >>= 1) vs += __shfl_xor(vs, off, 32);
  const float inv = rsqrtf(vs * (1.0f / 512.0f) + 1e-5f);
  float* po = out + (size_t)row * 512;
  __bf16* pob = outb + (size_t)row * 512;
#pragma unroll
  for (int i = 0; i < 16; ++i) {
    const int c = lane + 32 * i;
    const float y = (vals[i] - mu) * inv * g[c] + bta[c];
    po[c] = y;
    pob[c] = (__bf16)y;
  }
}

__global__ void row_sumsq(const float* __restrict__ Sig, float* __restrict__ sq) {
  const int i = blockIdx.x * blockDim.x + threadIdx.x;
  if (i >= 1024) return;
  const float* p = Sig + (size_t)i * 512;
  float s = 0.0f;
  for (int d = 0; d < 512; ++d) s += p[d] * p[d];
  sq[i] = s;
}

// f32 -> bf16 straight convert
__global__ void cvt_bf16(const float* __restrict__ in, __bf16* __restrict__ out, int n) {
  const int i = blockIdx.x * blockDim.x + threadIdx.x;
  if (i < n) out[i] = (__bf16)in[i];
}

// f32 [K,N] -> bf16 [N,K] transpose-convert (coalesced reads)
__global__ void transpose_cvt(const float* __restrict__ in, __bf16* __restrict__ out,
                              int Kd, int Nd) {
  const int i = blockIdx.x * blockDim.x + threadIdx.x;
  if (i >= Kd * Nd) return;
  const int k = i / Nd, n = i % Nd;
  out[(size_t)n * Kd + k] = (__bf16)in[i];
}

// top-4 routing: out = h + sum_k softmax(top4 logits)_k * Z[idx_k]  (bf16 out)
__global__ void route_topk(const float* __restrict__ logits, const float* __restrict__ h,
                           const float* __restrict__ Z, __bf16* __restrict__ out) {
  const int row = blockIdx.x * blockDim.x + threadIdx.x;
  if (row >= 8192) return;
  const float* lg = logits + (size_t)row * 1024;
  int idx[4]; float val[4];
#pragma unroll
  for (int t = 0; t < 4; ++t) {
    float best = -3.0e38f; int bi = 0;
    for (int j = 0; j < 1024; ++j) {
      bool used = false;
#pragma unroll
      for (int u = 0; u < 4; ++u) used |= (u < t) && (idx[u] == j);
      const float v = lg[j];
      if (!used && v > best) { best = v; bi = j; }
    }
    idx[t] = bi; val[t] = best;
  }
  float w[4]; float sum = 0.0f;
#pragma unroll
  for (int t = 0; t < 4; ++t) { w[t] = __expf(val[t] - val[0]); sum += w[t]; }
  const float rs = 1.0f / sum;
  const float* ph = h + (size_t)row * 512;
  __bf16* po = out + (size_t)row * 512;
  for (int d = 0; d < 512; ++d) {
    float a = ph[d];
#pragma unroll
    for (int t = 0; t < 4; ++t) a += w[t] * rs * Z[(size_t)idx[t] * 512 + d];
    po[d] = (__bf16)a;
  }
}

// ---------------------------------------------------------------------------

static inline unsigned cdiv(unsigned a, unsigned b) { return (a + b - 1) / b; }

extern "C" void kernel_launch(void* const* d_in, const int* in_sizes, int n_in,
                              void* d_out, int out_size, void* d_ws, size_t ws_size,
                              hipStream_t stream) {
  (void)in_sizes; (void)n_in; (void)out_size; (void)ws_size;
  const float* x_t    = (const float*)d_in[0];
  const float* t_emb  = (const float*)d_in[1];
  const float* Phi    = (const float*)d_in[2];
  const float* Sig    = (const float*)d_in[3];
  const float* Size_  = (const float*)d_in[4];
  /* d_in[5] q_ptrs unused */
  const float* w_in   = (const float*)d_in[6];
  const float* b_in   = (const float*)d_in[7];
  const float* attn_w = (const float*)d_in[8];
  const float* attn_b = (const float*)d_in[9];
  const float* out_w  = (const float*)d_in[10];
  const float* out_b  = (const float*)d_in[11];
  const float* ff1_w  = (const float*)d_in[12];
  const float* ff1_b  = (const float*)d_in[13];
  const float* ff2_w  = (const float*)d_in[14];
  const float* ff2_b  = (const float*)d_in[15];
  const float* ln1_g  = (const float*)d_in[16];
  const float* ln1_b  = (const float*)d_in[17];
  const float* ln2_g  = (const float*)d_in[18];
  const float* ln2_b  = (const float*)d_in[19];
  const float* ska_w[4] = {(const float*)d_in[20], (const float*)d_in[21],
                           (const float*)d_in[22], (const float*)d_in[23]};
  const float* wr     = (const float*)d_in[24];
  const float* w_out  = (const float*)d_in[25];
  const float* b_out  = (const float*)d_in[26];
  float* outp = (float*)d_out;

  // ---- workspace carving (256B aligned) ----
  char* wsb = (char*)d_ws;
  size_t off = 0;
  auto alloc = [&](size_t bytes) -> char* {
    char* p = wsb + off;
    off = (off + bytes + 255) & ~(size_t)255;
    return p;
  };
  const size_t T = 8192;  // B*S
  // f32 buffers
  float* H    = (float*)alloc(T * 512 * 4);
  float* PROJ = (float*)alloc(T * 512 * 4);
  float* PRI  = (float*)alloc((size_t)1024 * 1024 * 4);
  float* ZBf  = (float*)alloc((size_t)1024 * 512 * 4);
  float* LOG  = (float*)alloc(T * 1024 * 4);
  float* SQ   = (float*)alloc(1024 * 4);
  // bf16 activations
  __bf16* XB   = (__bf16*)alloc(T * 64 * 2);
  __bf16* HB   = (__bf16*)alloc(T * 512 * 2);
  __bf16* QKVB = (__bf16*)alloc(T * 1536 * 2);   // also router-mid (RB)
  __bf16* OB   = (__bf16*)alloc(T * 512 * 2);    // attn out / routed tokens
  __bf16* FB   = (__bf16*)alloc(T * 2048 * 2);
  __bf16* PhiB = (__bf16*)alloc((size_t)1024 * 512 * 2);
  __bf16* SigB = (__bf16*)alloc((size_t)1024 * 512 * 2);
  __bf16* BQB  = (__bf16*)alloc((size_t)1024 * 512 * 2);
  __bf16* BKB  = (__bf16*)alloc((size_t)1024 * 512 * 2);
  __bf16* BVB  = (__bf16*)alloc((size_t)1024 * 512 * 2);
  __bf16* ZPB  = (__bf16*)alloc((size_t)1024 * 512 * 2);
  // bf16 weights ([N,K] K-contiguous)
  __bf16* w_inT  = (__bf16*)alloc((size_t)512 * 64 * 2);
  __bf16* attnWB = (__bf16*)alloc((size_t)4 * 1536 * 512 * 2);  // already [N,K]
  __bf16* outWB  = (__bf16*)alloc((size_t)4 * 512 * 512 * 2);   // already [N,K]
  __bf16* ff1T   = (__bf16*)alloc((size_t)4 * 2048 * 512 * 2);
  __bf16* ff2T   = (__bf16*)alloc((size_t)4 * 512 * 2048 * 2);
  __bf16* skaT   = (__bf16*)alloc((size_t)4 * 512 * 512 * 2);   // q,k,v,o packed
  __bf16* wrT    = (__bf16*)alloc((size_t)512 * 512 * 2);
  __bf16* w_outT = (__bf16*)alloc((size_t)64 * 512 * 2);

  // ---- weight / input conversion passes ----
  cvt_bf16<<<cdiv(T * 64, 256), 256, 0, stream>>>(x_t, XB, T * 64);
  cvt_bf16<<<cdiv(4 * 1536 * 512, 256), 256, 0, stream>>>(attn_w, attnWB, 4 * 1536 * 512);
  cvt_bf16<<<cdiv(4 * 512 * 512, 256), 256, 0, stream>>>(out_w, outWB, 4 * 512 * 512);
  cvt_bf16<<<cdiv(1024 * 512, 256), 256, 0, stream>>>(Phi, PhiB, 1024 * 512);
  cvt_bf16<<<cdiv(1024 * 512, 256), 256, 0, stream>>>(Sig, SigB, 1024 * 512);
  transpose_cvt<<<cdiv(64 * 512, 256), 256, 0, stream>>>(w_in, w_inT, 64, 512);
  for (int l = 0; l < 4; ++l) {
    transpose_cvt<<<cdiv(512 * 2048, 256), 256, 0, stream>>>(
        ff1_w + (size_t)l * 512 * 2048, ff1T + (size_t)l * 2048 * 512, 512, 2048);
    transpose_cvt<<<cdiv(2048 * 512, 256), 256, 0, stream>>>(
        ff2_w + (size_t)l * 2048 * 512, ff2T + (size_t)l * 512 * 2048, 2048, 512);
    transpose_cvt<<<cdiv(512 * 512, 256), 256, 0, stream>>>(
        ska_w[l], skaT + (size_t)l * 512 * 512, 512, 512);
  }
  transpose_cvt<<<cdiv(512 * 512, 256), 256, 0, stream>>>(wr, wrT, 512, 512);
  transpose_cvt<<<cdiv(512 * 64, 256), 256, 0, stream>>>(w_out, w_outT, 512, 64);

  // 1) h = x_t @ w_in + b_in + PE + t_embed  (dual f32 + bf16)
  gemm_bf16_wmma<3, 1, 1><<<dim3((T / 16) * 32 / 4), 128, 0, stream>>>(
      XB, w_inT, b_in, H, HB, 32, 64, 64, 64, 512, 1.0f, t_emb, nullptr);

  // 2) transformer layers
  for (int l = 0; l < 4; ++l) {
    gemm_bf16_wmma<1, 0, 1><<<dim3((T / 16) * 96 / 4), 128, 0, stream>>>(
        HB, attnWB + (size_t)l * 1536 * 512, attn_b + l * 1536, nullptr, QKVB,
        96, 512, 512, 512, 1536, 1.0f, nullptr, nullptr);
    token_attn<<<dim3(16 * 8 * 32), 32, 0, stream>>>(QKVB, OB);
    gemm_bf16_wmma<1, 1, 0><<<dim3((T / 16) * 32 / 4), 128, 0, stream>>>(
        OB, outWB + (size_t)l * 512 * 512, out_b + l * 512, PROJ, nullptr,
        32, 512, 512, 512, 512, 1.0f, nullptr, nullptr);
    residual_ln<<<dim3(1024), 256, 0, stream>>>(H, PROJ, ln1_g + l * 512, ln1_b + l * 512, H, HB);
    gemm_bf16_wmma<2, 0, 1><<<dim3((T / 16) * 128 / 4), 128, 0, stream>>>(
        HB, ff1T + (size_t)l * 2048 * 512, ff1_b + l * 2048, nullptr, FB,
        128, 512, 512, 512, 2048, 1.0f, nullptr, nullptr);
    gemm_bf16_wmma<1, 1, 0><<<dim3((T / 16) * 32 / 4), 128, 0, stream>>>(
        FB, ff2T + (size_t)l * 512 * 2048, ff2_b + l * 512, PROJ, nullptr,
        32, 2048, 2048, 2048, 512, 1.0f, nullptr, nullptr);
    residual_ln<<<dim3(1024), 256, 0, stream>>>(H, PROJ, ln2_g + l * 512, ln2_b + l * 512, H, HB);
  }

  // 3) bank attention
  __bf16* bqkv[3] = {BQB, BKB, BVB};
  for (int i = 0; i < 3; ++i)
    gemm_bf16_wmma<0, 0, 1><<<dim3(64 * 32 / 4), 128, 0, stream>>>(
        PhiB, skaT + (size_t)i * 512 * 512, nullptr, nullptr, bqkv[i],
        32, 512, 512, 512, 512, 1.0f, nullptr, nullptr);
  row_sumsq<<<dim3(4), 256, 0, stream>>>(Sig, SQ);
  gemm_bf16_wmma<5, 1, 0><<<dim3(64 * 64 / 4), 128, 0, stream>>>(
      SigB, SigB, nullptr, PRI, nullptr, 64, 512, 512, 512, 1024, 1.0f, SQ, Size_);
  bank_attn<<<dim3(8 * 64), 32, 0, stream>>>(BQB, BKB, BVB, PRI, ZPB);
  gemm_bf16_wmma<0, 1, 0><<<dim3(64 * 32 / 4), 128, 0, stream>>>(
      ZPB, skaT + (size_t)3 * 512 * 512, nullptr, ZBf, nullptr,
      32, 512, 512, 512, 512, 1.0f, nullptr, nullptr);

  // 4) router + output projection
  gemm_bf16_wmma<0, 0, 1><<<dim3((T / 16) * 32 / 4), 128, 0, stream>>>(
      HB, wrT, nullptr, nullptr, QKVB, 32, 512, 512, 512, 512, 1.0f, nullptr, nullptr);
  gemm_bf16_wmma<4, 1, 0><<<dim3((T / 16) * 64 / 4), 128, 0, stream>>>(
      QKVB, PhiB, nullptr, LOG, nullptr, 64, 512, 512, 512, 1024,
      0.044194174f, nullptr, nullptr);  // 1/sqrt(512)
  route_topk<<<dim3(32), 256, 0, stream>>>(LOG, H, ZBf, OB);
  gemm_bf16_wmma<1, 1, 0><<<dim3((T / 16) * 4 / 4), 128, 0, stream>>>(
      OB, w_outT, b_out, outp, nullptr, 4, 512, 512, 512, 64, 1.0f, nullptr, nullptr);
}
